// Glm4MoeSparseMoeBlock_2491081031867
// MI455X (gfx1250) — matrix operand order repaired and checked by reading, not compiled, not run on hardware
//
#include <hip/hip_runtime.h>

// ---------------- problem constants ----------------
#define Tt   2048
#define Hd   1024
#define Id   512        // I (= Is for shared expert)
#define Ed   32
#define Kk   8
#define NGRP 4
#define RSCALE 2.5f
#define LDSH_STRIDE 520 // 512 + 8 ushorts pad -> conflict-free ds_load_b128

// TDM staging geometry: per K-chunk panel = 128 rows x 64B, padded to 80B/row by TDM
#define ROW_U   40                  // ushorts per staged row (64B data + 16B pad)
#define PANEL_B 10240               // bytes per panel (128 * 80)
#define PANEL_U 5120                // ushorts per panel
#define STAGE_BYTES (8 * 2 * PANEL_B)           // 8 waves, double buffered = 160 KB
#define LDSH_OFF    STAGE_BYTES
#define LDSH_BYTES  (32 * LDSH_STRIDE * 2)      // 33,280
#define TOK_OFF     (LDSH_OFF + LDSH_BYTES)
#define WGT_OFF     (TOK_OFF + 128)
#define SMEM_TOTAL  (WGT_OFF + 128)             // ~193.5 KB (<320 KB/WG per ISA)

// ---------------- WMMA / TDM types ----------------
typedef __bf16 bf16_t;
typedef bf16_t v16bf __attribute__((ext_vector_type(16)));
typedef bf16_t v8bf  __attribute__((ext_vector_type(8)));
typedef float  v8f   __attribute__((ext_vector_type(8)));
typedef unsigned int u32x4 __attribute__((ext_vector_type(4)));
typedef int          i32x8 __attribute__((ext_vector_type(8)));
typedef int          i32x4 __attribute__((ext_vector_type(4)));

union FragB { v16bf v; v8bf h[2]; };

__device__ __forceinline__ v8f vzero8() {
  v8f z = {0.f,0.f,0.f,0.f,0.f,0.f,0.f,0.f};
  return z;
}

__device__ __forceinline__ unsigned short f32_to_bf16(float f) {
  unsigned int u = __float_as_uint(f);
  unsigned int r = u + 0x7FFFu + ((u >> 16) & 1u);   // round-to-nearest-even
  return (unsigned short)(r >> 16);
}

// Tensor Data Mover: async global->LDS tile load (TENSORcnt).
// 3D tile [32 x tile_d1 x tile_d2] of bf16; rows 64B, padded to 80B in LDS
// (pad_interval=16 DW, pad_amount=4 DW) for alignment + bank rotation.
__device__ __forceinline__ void tdm_load(const unsigned short* gsrc, unsigned lds_off,
                                         unsigned d0_stride_elems, unsigned tile_d1,
                                         unsigned tile_d2, unsigned d1_stride_elems) {
  unsigned long long ga = (unsigned long long)(uintptr_t)gsrc;
  u32x4 g0 = {0u, 0u, 0u, 0u};
  g0[0] = 1u;                                             // count=1 (valid user D#)
  g0[1] = lds_off;                                        // lds_addr (bytes)
  g0[2] = (unsigned)ga;                                   // global_addr[31:0]
  g0[3] = (unsigned)((ga >> 32) & 0x01FFFFFFull) | (2u << 30);  // ga[56:32] | type=2
  i32x8 g1 = {0,0,0,0,0,0,0,0};
  // wg_mask=0 | data_size=1(2B) | pad_enable | pad_interval=3(16DW) | pad_amount=3(4DW)
  g1[0] = (int)((1u << 16) | (1u << 20) | (3u << 22) | (3u << 25));
  g1[1] = 0;                                              // atomic_barrier_addr=0, td0 lo
  g1[2] = (int)0x00004000u;                               // tensor_dim0 = 2^30 (no OOB)
  g1[3] = (int)(0x00004000u | (32u << 16));               // tensor_dim1 = 2^30, tile_dim0=32
  g1[4] = (int)(tile_d1 | (tile_d2 << 16));               // tile_dim1, tile_dim2
  g1[5] = (int)d0_stride_elems;                           // tensor_dim0_stride[31:0]
  g1[6] = (int)((d1_stride_elems & 0xFFFFu) << 16);       // dim1_stride[15:0]
  g1[7] = (int)(d1_stride_elems >> 16);                   // dim1_stride[47:16]
  i32x4 g2 = {(int)0x40000000, 0, 0, 0};                  // tensor_dim2 = 2^30
  i32x4 g3 = {0, 0, 0, 0};
  i32x8 g4 = {0, 0, 0, 0, 0, 0, 0, 0};                    // unused extra group (clang-23 form)
  __builtin_amdgcn_tensor_load_to_lds(g0, g1, g2, g3, g4, 0);
}

// ---------------- fp32 -> bf16 conversion (grid-stride, 4-wide) ----------------
__global__ __launch_bounds__(256)
void cvt_kernel(const float* __restrict__ src, unsigned short* __restrict__ dst, int n4) {
  int i = blockIdx.x * blockDim.x + threadIdx.x;
  int stride = gridDim.x * blockDim.x;
  const float4* s4 = (const float4*)src;
  ushort4* d4 = (ushort4*)dst;
  for (; i < n4; i += stride) {
    float4 f = s4[i];
    ushort4 o;
    o.x = f32_to_bf16(f.x); o.y = f32_to_bf16(f.y);
    o.z = f32_to_bf16(f.z); o.w = f32_to_bf16(f.w);
    d4[i] = o;
  }
}

// ---------------- router: one wave (32 lanes = 32 experts) per token ----------------
__global__ __launch_bounds__(256)
void router_kernel(const float* __restrict__ x, const float* __restrict__ gate_w,
                   const float* __restrict__ bias, float* __restrict__ dense_w) {
  __shared__ float lds_x[8][Hd];
  __shared__ float s_sc[8][Ed];
  __shared__ float s_sfc[8][Ed];
  __shared__ float s_wsel[8][Ed];
  int tid = threadIdx.x, lane = tid & 31, wv = tid >> 5;
  int t = blockIdx.x * 8 + wv;
  const float* xr = x + (size_t)t * Hd;
  for (int j = lane; j < Hd; j += 32) lds_x[wv][j] = xr[j];
  s_wsel[wv][lane] = 0.f;
  __syncthreads();

  const float* gr = gate_w + (size_t)lane * Hd;
  float acc = 0.f;
  for (int h = 0; h < Hd; h += 4) {
    acc += lds_x[wv][h+0] * gr[h+0];
    acc += lds_x[wv][h+1] * gr[h+1];
    acc += lds_x[wv][h+2] * gr[h+2];
    acc += lds_x[wv][h+3] * gr[h+3];
  }
  float score = 1.0f / (1.0f + __expf(-acc));
  s_sc[wv][lane]  = score;
  s_sfc[wv][lane] = score + bias[lane];
  __syncthreads();

  if (lane == 0) {
    float gsum[NGRP];
    for (int g = 0; g < NGRP; g++) {
      float m1 = -3.4e38f; int i1 = -1;
      for (int j = 0; j < 8; j++) { float v = s_sfc[wv][g*8+j]; if (v > m1) { m1 = v; i1 = j; } }
      float m2 = -3.4e38f;
      for (int j = 0; j < 8; j++) { if (j != i1) { float v = s_sfc[wv][g*8+j]; if (v > m2) m2 = v; } }
      gsum[g] = m1 + m2;
    }
    int g1 = 0; { float b = gsum[0]; for (int g = 1; g < NGRP; g++) if (gsum[g] > b) { b = gsum[g]; g1 = g; } }
    int g2 = -1; { float b = -3.4e38f; for (int g = 0; g < NGRP; g++) if (g != g1 && gsum[g] > b) { b = gsum[g]; g2 = g; } }
    unsigned taken = 0;
    int sel[Kk]; float wsum = 0.f;
    for (int k = 0; k < Kk; k++) {
      float b = -3.4e38f; int bi = 0;
      for (int e = 0; e < Ed; e++) {
        int g = e >> 3;
        if (g != g1 && g != g2) continue;
        if (taken & (1u << e)) continue;
        float v = s_sfc[wv][e];
        if (v > b) { b = v; bi = e; }
      }
      taken |= 1u << bi; sel[k] = bi; wsum += s_sc[wv][bi];
    }
    for (int k = 0; k < Kk; k++)
      s_wsel[wv][sel[k]] = s_sc[wv][sel[k]] / wsum * RSCALE;
  }
  __syncthreads();
  dense_w[(size_t)t * Ed + lane] = s_wsel[wv][lane];
}

// ---------------- deterministic per-expert token lists (lane = expert) ----------------
__global__ void build_lists_kernel(const float* __restrict__ dense_w,
                                   int* __restrict__ cnt, int* __restrict__ tlist,
                                   float* __restrict__ wlist) {
  int e = threadIdx.x;
  if (e >= Ed) return;
  int c = 0;
  for (int t = 0; t < Tt; t++) {
    float w = dense_w[(size_t)t * Ed + e];
    if (w != 0.f) { tlist[e * Tt + c] = t; wlist[e * Tt + c] = w; c++; }
  }
  cnt[e] = c;
}

// ---------------- fused expert FFN with TDM-staged weights ----------------
// Block = 256 threads (8 waves), one 32-token tile of one expert.
// Weights stream via tensor_load_to_lds into per-wave double buffers (TENSORcnt
// pipelined: issue chunk k+1, s_wait_tensorcnt 1, compute chunk k from LDS).
__global__ __launch_bounds__(256)
void ffn_kernel(const unsigned short* __restrict__ xb,   // [T,H] bf16
                const unsigned short* __restrict__ wgu,  // [E,2I,H] or [2I,H] bf16
                const unsigned short* __restrict__ wd,   // [E,H,I]  or [H,I]  bf16
                const int* __restrict__ tlist, const float* __restrict__ wlist,
                const int* __restrict__ cnt,
                float* __restrict__ out, int shared_mode) {
  extern __shared__ unsigned char smem[];
  unsigned short* stage = (unsigned short*)smem;
  unsigned short* lds_h = (unsigned short*)(smem + LDSH_OFF);
  int*   s_tok = (int*)(smem + TOK_OFF);
  float* s_w   = (float*)(smem + WGT_OFF);
  unsigned smem_base = (unsigned)(uintptr_t)smem;

  int e = blockIdx.y;
  int tile = blockIdx.x;
  const unsigned short* Wgu;
  const unsigned short* Wd;
  if (shared_mode) {
    Wgu = wgu; Wd = wd;
    if (threadIdx.x < 32) {
      s_tok[threadIdx.x] = tile * 32 + threadIdx.x;
      s_w[threadIdx.x] = 1.0f;
    }
  } else {
    int c = cnt[e];
    if (tile * 32 >= c) return;
    int n_tok = c - tile * 32; if (n_tok > 32) n_tok = 32;
    Wgu = wgu + (size_t)e * (2 * Id) * Hd;
    Wd  = wd  + (size_t)e * Hd * Id;
    if (threadIdx.x < 32) {
      int m = threadIdx.x;
      if (m < n_tok) { s_tok[m] = tlist[e * Tt + tile * 32 + m]; s_w[m] = wlist[e * Tt + tile * 32 + m]; }
      else           { s_tok[m] = tlist[e * Tt + tile * 32];     s_w[m] = 0.0f; }
    }
  }
  __syncthreads();

  int tid = threadIdx.x, lane = tid & 31, wv = tid >> 5;
  int m    = lane & 15;        // A row / B column index within M-tile
  int hi   = lane >> 4;        // lane half (K split per ISA layout)
  int koff = hi ? 8 : 0;
  const unsigned short* arow0 = xb + (size_t)s_tok[m]      * Hd;
  const unsigned short* arow1 = xb + (size_t)s_tok[16 + m] * Hd;

  unsigned stage_off0 = smem_base + (unsigned)(wv * 2 * PANEL_B);
  unsigned stage_off1 = stage_off0 + PANEL_B;
  const unsigned short* sbuf0 = stage + (size_t)(wv * 2)     * PANEL_U;
  const unsigned short* sbuf1 = stage + (size_t)(wv * 2 + 1) * PANEL_U;

  // ---- phase 1: gate_up = X @ Wgu^T ----
  v8f accg[2][4], accu[2][4];
  for (int mt = 0; mt < 2; mt++)
    for (int i = 0; i < 4; i++) { accg[mt][i] = vzero8(); accu[mt][i] = vzero8(); }
  int og = wv * 64;
  const unsigned short* gbase1 = Wgu + (size_t)og * Hd;

  auto compute_p1 = [&](int kc, const unsigned short* sb) {
    int kb = kc * 32;
    FragB a0, a1;
    a0.h[0] = *(const v8bf*)(arow0 + kb + koff);
    a0.h[1] = *(const v8bf*)(arow0 + kb + 16 + koff);
    a1.h[0] = *(const v8bf*)(arow1 + kb + koff);
    a1.h[1] = *(const v8bf*)(arow1 + kb + 16 + koff);
    if (kb + 32 < Hd) {
      __builtin_prefetch(arow0 + kb + 32, 0, 1);
      __builtin_prefetch(arow1 + kb + 32, 0, 1);
    }
    for (int nt = 0; nt < 4; nt++) {
      const unsigned short* bgp = sb + (nt * 16 + m) * ROW_U + hi * 8;
      FragB bg;
      bg.h[0] = *(const v8bf*)(bgp);
      bg.h[1] = *(const v8bf*)(bgp + 16);
      accg[0][nt] = __builtin_amdgcn_wmma_f32_16x16x32_bf16(false, a0.v, false, bg.v,
                                                            (short)0, accg[0][nt], false, false);
      accg[1][nt] = __builtin_amdgcn_wmma_f32_16x16x32_bf16(false, a1.v, false, bg.v,
                                                            (short)0, accg[1][nt], false, false);
      const unsigned short* bup = sb + (64 + nt * 16 + m) * ROW_U + hi * 8;
      FragB bu;
      bu.h[0] = *(const v8bf*)(bup);
      bu.h[1] = *(const v8bf*)(bup + 16);
      accu[0][nt] = __builtin_amdgcn_wmma_f32_16x16x32_bf16(false, a0.v, false, bu.v,
                                                            (short)0, accu[0][nt], false, false);
      accu[1][nt] = __builtin_amdgcn_wmma_f32_16x16x32_bf16(false, a1.v, false, bu.v,
                                                            (short)0, accu[1][nt], false, false);
    }
  };

  // pipelined TDM double buffer: 3D tile [32 x 64 x 2] (gate half + up half)
  tdm_load(gbase1, stage_off0, Hd, 64, 2, (unsigned)(Id * Hd));
  for (int kc = 0; kc < 31; kc++) {
    tdm_load(gbase1 + (kc + 1) * 32, ((kc + 1) & 1) ? stage_off1 : stage_off0,
             Hd, 64, 2, (unsigned)(Id * Hd));
    __builtin_amdgcn_s_wait_tensorcnt((short)1);
    compute_p1(kc, (kc & 1) ? sbuf1 : sbuf0);
  }
  __builtin_amdgcn_s_wait_tensorcnt((short)0);
  compute_p1(31, sbuf1);

  // in-register SwiGLU (+ routing weight), write bf16 h to LDS
  for (int mt = 0; mt < 2; mt++) {
    for (int nt = 0; nt < 4; nt++) {
      int i0 = og + nt * 16 + m;
      for (int v = 0; v < 8; v++) {
        int row = mt * 16 + v + 8 * hi;
        float g = accg[mt][nt][v], u = accu[mt][nt][v];
        float hval = (g / (1.0f + __expf(-g))) * u * s_w[row];
        lds_h[row * LDSH_STRIDE + i0] = f32_to_bf16(hval);
      }
    }
  }

  // ---- phase 2: out += h @ Wd^T ----
  v8f acc2[2][8];
  for (int mt = 0; mt < 2; mt++)
    for (int i = 0; i < 8; i++) acc2[mt][i] = vzero8();
  int ob = wv * 128;
  const unsigned short* gbase2 = Wd + (size_t)ob * Id;

  // prologue TDM touches only this wave's staging (safe pre-barrier);
  // barrier orders all waves' lds_h writes before phase-2 A reads.
  tdm_load(gbase2, stage_off0, Id, 128, 0, 0);
  __syncthreads();

  auto compute_p2 = [&](int kc, const unsigned short* sb) {
    int kb = kc * 32;
    FragB a0, a1;
    a0.h[0] = *(const v8bf*)(&lds_h[m * LDSH_STRIDE + kb + koff]);
    a0.h[1] = *(const v8bf*)(&lds_h[m * LDSH_STRIDE + kb + 16 + koff]);
    a1.h[0] = *(const v8bf*)(&lds_h[(16 + m) * LDSH_STRIDE + kb + koff]);
    a1.h[1] = *(const v8bf*)(&lds_h[(16 + m) * LDSH_STRIDE + kb + 16 + koff]);
    for (int nt = 0; nt < 8; nt++) {
      const unsigned short* bp = sb + (nt * 16 + m) * ROW_U + hi * 8;
      FragB b;
      b.h[0] = *(const v8bf*)(bp);
      b.h[1] = *(const v8bf*)(bp + 16);
      acc2[0][nt] = __builtin_amdgcn_wmma_f32_16x16x32_bf16(false, a0.v, false, b.v,
                                                            (short)0, acc2[0][nt], false, false);
      acc2[1][nt] = __builtin_amdgcn_wmma_f32_16x16x32_bf16(false, a1.v, false, b.v,
                                                            (short)0, acc2[1][nt], false, false);
    }
  };

  for (int kc = 0; kc < 15; kc++) {
    tdm_load(gbase2 + (kc + 1) * 32, ((kc + 1) & 1) ? stage_off1 : stage_off0,
             Id, 128, 0, 0);
    __builtin_amdgcn_s_wait_tensorcnt((short)1);
    compute_p2(kc, (kc & 1) ? sbuf1 : sbuf0);
  }
  __builtin_amdgcn_s_wait_tensorcnt((short)0);
  compute_p2(15, sbuf1);

  for (int mt = 0; mt < 2; mt++) {
    for (int nt = 0; nt < 8; nt++) {
      for (int v = 0; v < 8; v++) {
        int row = mt * 16 + v + 8 * hi;
        size_t idx = (size_t)s_tok[row] * Hd + ob + nt * 16 + m;
        if (shared_mode) out[idx] = acc2[mt][nt][v];
        else             atomicAdd(&out[idx], acc2[mt][nt][v]);
      }
    }
  }
}

// ---------------- host-side launch ----------------
extern "C" void kernel_launch(void* const* d_in, const int* in_sizes, int n_in,
                              void* d_out, int out_size, void* d_ws, size_t ws_size,
                              hipStream_t stream) {
  const float* x      = (const float*)d_in[0];   // [T,H]
  const float* gate_w = (const float*)d_in[1];   // [E,H]
  const float* bias   = (const float*)d_in[2];   // [E]
  const float* wgu    = (const float*)d_in[3];   // [E,2I,H]
  const float* wdn    = (const float*)d_in[4];   // [E,H,I]
  const float* wsgu   = (const float*)d_in[5];   // [2I,H]
  const float* wsdn   = (const float*)d_in[6];   // [H,I]
  float* out = (float*)d_out;

  const size_t n_x    = (size_t)Tt * Hd;
  const size_t n_wgu  = (size_t)Ed * 2 * Id * Hd;
  const size_t n_wd   = (size_t)Ed * Hd * Id;
  const size_t n_wsgu = (size_t)2 * Id * Hd;
  const size_t n_wsd  = (size_t)Hd * Id;

  char* base = (char*)d_ws;
  size_t off = 0;
  auto carve = [&](size_t bytes) -> char* {
    char* p = base + off;
    off += (bytes + 255) & ~(size_t)255;
    return p;
  };
  unsigned short* xb    = (unsigned short*)carve(n_x * 2);
  unsigned short* wgub  = (unsigned short*)carve(n_wgu * 2);
  unsigned short* wdb   = (unsigned short*)carve(n_wd * 2);
  unsigned short* wsgub = (unsigned short*)carve(n_wsgu * 2);
  unsigned short* wsdb  = (unsigned short*)carve(n_wsd * 2);
  float* dense_w = (float*)carve((size_t)Tt * Ed * 4);
  int*   cnt     = (int*)carve(Ed * 4);
  int*   tlist   = (int*)carve((size_t)Ed * Tt * 4);
  float* wlist   = (float*)carve((size_t)Ed * Tt * 4);

  auto cvt = [&](const float* s, unsigned short* d, size_t n) {
    int n4 = (int)(n / 4);
    int grid = (n4 + 255) / 256; if (grid > 8192) grid = 8192;
    cvt_kernel<<<grid, 256, 0, stream>>>(s, d, n4);
  };
  cvt(x, xb, n_x);
  cvt(wgu, wgub, n_wgu);
  cvt(wdn, wdb, n_wd);
  cvt(wsgu, wsgub, n_wsgu);
  cvt(wsdn, wsdb, n_wsd);

  router_kernel<<<Tt / 8, 256, 0, stream>>>(x, gate_w, bias, dense_w);
  build_lists_kernel<<<1, 32, 0, stream>>>(dense_w, cnt, tlist, wlist);

  // shared expert first: plain stores initialize d_out
  ffn_kernel<<<dim3(Tt / 32, 1), 256, SMEM_TOTAL, stream>>>(xb, wsgub, wsdb,
                                                            nullptr, nullptr, nullptr, out, 1);
  // routed experts: grouped GEMM, atomic accumulate
  ffn_kernel<<<dim3(Tt / 32, Ed), 256, SMEM_TOTAL, stream>>>(xb, wgub, wdb,
                                                             tlist, wlist, cnt, out, 0);
}
